// Int8Linear_42554535969583
// MI455X (gfx1250) — 8-bit matrix operand order repaired and checked by reading, compiled, not run on hardware
//
#include <hip/hip_runtime.h>
#include <hip/hip_bf16.h>

typedef signed char i8;
typedef __attribute__((ext_vector_type(8))) int v8i;
typedef __attribute__((ext_vector_type(4))) int v4i;

#define BLK 128
#define EPS 1e-12f

#if __has_builtin(__builtin_amdgcn_global_load_async_to_lds_b128) && \
    __has_builtin(__builtin_amdgcn_s_wait_asynccnt)
#define USE_ASYNC_LDS 1
typedef __attribute__((address_space(1))) v4i* gptr_b128;
typedef __attribute__((address_space(3))) v4i* lptr_b128;
#else
#define USE_ASYNC_LDS 0
#endif

__global__ __launch_bounds__(256) void quant_x_kernel(
    const float* __restrict__ x, i8* __restrict__ xq, float* __restrict__ xs,
    int M, int K) {
  const int Kb   = K >> 7;
  const int warp = (blockIdx.x * blockDim.x + threadIdx.x) >> 5;
  const int lane = threadIdx.x & 31;
  if (warp >= M * Kb) return;
  const int m  = warp / Kb;
  const int kb = warp % Kb;

  const float4* p = (const float4*)(x + (size_t)m * K + (size_t)kb * BLK);
  float4 v = p[lane];
  float a = fmaxf(fmaxf(fabsf(v.x), fabsf(v.y)), fmaxf(fabsf(v.z), fabsf(v.w)));
#pragma unroll
  for (int off = 16; off > 0; off >>= 1) a = fmaxf(a, __shfl_xor(a, off, 32));
  const float s = fmaxf(a * (1.0f / 127.0f), EPS);
  if (lane == 0) xs[(size_t)m * Kb + kb] = s;

  int q0 = (int)rintf(fminf(fmaxf(v.x / s, -127.0f), 127.0f));
  int q1 = (int)rintf(fminf(fmaxf(v.y / s, -127.0f), 127.0f));
  int q2 = (int)rintf(fminf(fmaxf(v.z / s, -127.0f), 127.0f));
  int q3 = (int)rintf(fminf(fmaxf(v.w / s, -127.0f), 127.0f));
  int packed = (q0 & 255) | ((q1 & 255) << 8) | ((q2 & 255) << 16) | ((q3 & 255) << 24);
  ((int*)(xq + (size_t)m * K + (size_t)kb * BLK))[lane] = packed;
}

__global__ __launch_bounds__(256) void quant_w_kernel(
    const float* __restrict__ w, i8* __restrict__ wq, float* __restrict__ ws,
    int N, int K) {
  const int Kb  = K >> 7;
  const int nb  = blockIdx.x / Kb;
  const int kb  = blockIdx.x % Kb;
  const int tid = threadIdx.x;
  const int row  = nb * BLK + (tid >> 1);
  const int colb = kb * BLK + (tid & 1) * 64;

  const float4* p = (const float4*)(w + (size_t)row * K + colb);
  float a = 0.0f;
#pragma unroll
  for (int i = 0; i < 16; ++i) {
    float4 v = p[i];
    a = fmaxf(a, fmaxf(fmaxf(fabsf(v.x), fabsf(v.y)), fmaxf(fabsf(v.z), fabsf(v.w))));
  }
#pragma unroll
  for (int off = 16; off > 0; off >>= 1) a = fmaxf(a, __shfl_xor(a, off, 32));

  __shared__ float red[8];
  __shared__ float sS;
  if ((tid & 31) == 0) red[tid >> 5] = a;
  __syncthreads();
  if (tid == 0) {
    float m = red[0];
#pragma unroll
    for (int i = 1; i < 8; ++i) m = fmaxf(m, red[i]);
    float s = fmaxf(m * (1.0f / 127.0f), EPS);
    sS = s;
    ws[(size_t)nb * Kb + kb] = s;
  }
  __syncthreads();
  const float s = sS;

  int* q = (int*)(wq + (size_t)row * K + colb);
#pragma unroll
  for (int i = 0; i < 16; ++i) {
    float4 v = p[i];
    int q0 = (int)rintf(fminf(fmaxf(v.x / s, -127.0f), 127.0f));
    int q1 = (int)rintf(fminf(fmaxf(v.y / s, -127.0f), 127.0f));
    int q2 = (int)rintf(fminf(fmaxf(v.z / s, -127.0f), 127.0f));
    int q3 = (int)rintf(fminf(fmaxf(v.w / s, -127.0f), 127.0f));
    q[i] = (q0 & 255) | ((q1 & 255) << 8) | ((q2 & 255) << 16) | ((q3 & 255) << 24);
  }
}

__global__ __launch_bounds__(256) void gemm_i8_kernel(
    const i8* __restrict__ xq, const float* __restrict__ xs,
    const i8* __restrict__ wq, const float* __restrict__ wsc,
    const float* __restrict__ bias, float* __restrict__ out,
    int M, int N, int K) {
  const int Kb = K >> 7;

  __shared__ i8    sA[BLK * BLK];
  __shared__ i8    sB[BLK * BLK];
  __shared__ float sXs[BLK];

  const int tid  = threadIdx.x;
  const int lane = tid & 31;
  const int wave = tid >> 5;
  const int wm   = wave & 3;
  const int wn   = wave >> 2;
  const int l16  = lane & 15;
  const int half = lane >> 4;

  const int m0 = blockIdx.x * BLK;
  const int n0 = blockIdx.y * BLK;

#if USE_ASYNC_LDS
  i8* xq_nc = const_cast<i8*>(xq);
  i8* wq_nc = const_cast<i8*>(wq);
#endif

  float facc[2][4][8];
#pragma unroll
  for (int ti = 0; ti < 2; ++ti)
#pragma unroll
    for (int tj = 0; tj < 4; ++tj)
#pragma unroll
      for (int r = 0; r < 8; ++r) facc[ti][tj][r] = 0.0f;

  for (int kb = 0; kb < Kb; ++kb) {
    const int k0 = kb << 7;
    __syncthreads();
#pragma unroll
    for (int p = 0; p < 4; ++p) {
      int c   = p * 256 + tid;
      int row = c >> 3;
      int col = c & 7;
#if USE_ASYNC_LDS
      __builtin_amdgcn_global_load_async_to_lds_b128(
          (gptr_b128)(xq_nc + (size_t)(m0 + row) * K + k0 + col * 16),
          (lptr_b128)(sA + c * 16), 0, 0);
      __builtin_amdgcn_global_load_async_to_lds_b128(
          (gptr_b128)(wq_nc + (size_t)(n0 + row) * K + k0 + col * 16),
          (lptr_b128)(sB + c * 16), 0, 0);
#else
      ((int4*)sA)[c] = *((const int4*)(xq + (size_t)(m0 + row) * K + k0) + col);
      ((int4*)sB)[c] = *((const int4*)(wq + (size_t)(n0 + row) * K + k0) + col);
#endif
    }
    if (tid < BLK) sXs[tid] = xs[(size_t)(m0 + tid) * Kb + kb];
    if (kb + 1 < Kb) {
      __builtin_prefetch(xq + (size_t)(m0 + (tid >> 1)) * K + k0 + BLK + (tid & 1) * 64, 0, 1);
      __builtin_prefetch(wq + (size_t)(n0 + (tid >> 1)) * K + k0 + BLK + (tid & 1) * 64, 0, 1);
    }
#if USE_ASYNC_LDS
    __builtin_amdgcn_s_wait_asynccnt(0);
#endif
    __syncthreads();

    const float wsv = wsc[(size_t)(n0 >> 7) * Kb + kb];
    float xscale[2][8];
#pragma unroll
    for (int ti = 0; ti < 2; ++ti)
#pragma unroll
      for (int r = 0; r < 8; ++r)
        xscale[ti][r] = sXs[wm * 32 + ti * 16 + r + half * 8] * wsv;

    const int* sA32 = (const int*)sA;
    const int* sB32 = (const int*)sB;

    v8i iacc[2][4];
#pragma unroll
    for (int kk = 0; kk < 2; ++kk) {
      v8i afrag[2];
      v8i bfrag[4];
#pragma unroll
      for (int ti = 0; ti < 2; ++ti) {
        int mrow = wm * 32 + ti * 16 + l16;
        int base = mrow * 32 + kk * 16;
#pragma unroll
        for (int d = 0; d < 8; ++d) {
          int w4   = d & 3;
          int koff = ((d >= 4) ? 8 : 0) + half * 4 + w4;
          afrag[ti][d] = sA32[base + koff];
        }
      }
#pragma unroll
      for (int tj = 0; tj < 4; ++tj) {
        int nrow = wn * 64 + tj * 16 + l16;
        int base = nrow * 32 + kk * 16;
#pragma unroll
        for (int d = 0; d < 8; ++d) {
          int koff = ((d >= 4) ? 8 : 0) + half * 4 + (d & 3);
          bfrag[tj][d] = sB32[base + koff];
        }
      }
#pragma unroll
      for (int ti = 0; ti < 2; ++ti) {
#pragma unroll
        for (int tj = 0; tj < 4; ++tj) {
          v8i cin;
          if (kk == 0) {
            v8i z = {};
            cin = z;
          } else {
            cin = iacc[ti][tj];
          }
          iacc[ti][tj] = __builtin_amdgcn_wmma_i32_16x16x64_iu8(
               true, afrag[ti],  true, bfrag[tj],
              cin,  false,  false);
        }
      }
    }
#pragma unroll
    for (int ti = 0; ti < 2; ++ti)
#pragma unroll
      for (int tj = 0; tj < 4; ++tj)
#pragma unroll
        for (int r = 0; r < 8; ++r)
          facc[ti][tj][r] += (float)iacc[ti][tj][r] * xscale[ti][r];
  }

#pragma unroll
  for (int tj = 0; tj < 4; ++tj) {
    int n = n0 + wn * 64 + tj * 16 + l16;
    float bv = bias[n];
#pragma unroll
    for (int ti = 0; ti < 2; ++ti) {
#pragma unroll
      for (int r = 0; r < 8; ++r) {
        int m = m0 + wm * 32 + ti * 16 + r + half * 8;
        out[(size_t)m * N + n] = facc[ti][tj][r] + bv;
      }
    }
  }
}

extern "C" void kernel_launch(void* const* d_in, const int* in_sizes, int n_in,
                              void* d_out, int out_size, void* d_ws, size_t ws_size,
                              hipStream_t stream) {
  const float* x    = (const float*)d_in[0];
  const float* w    = (const float*)d_in[1];
  const float* bias = (const float*)d_in[2];
  float* out = (float*)d_out;

  const int N = in_sizes[2];
  const int K = in_sizes[1] / N;
  const int M = in_sizes[0] / K;
  const int Kb = K / BLK;
  const int Nb = N / BLK;

  char* ws = (char*)d_ws;
  i8*    xq  = (i8*)ws;
  float* xs  = (float*)(ws + (size_t)M * K);
  i8*    wq  = (i8*)(ws + (size_t)M * K + (size_t)M * Kb * sizeof(float));
  float* wsc = (float*)(ws + (size_t)M * K + (size_t)M * Kb * sizeof(float) +
                        (size_t)N * K);

  {
    int warps  = M * Kb;
    int blocks = (warps * 32 + 255) / 256;
    quant_x_kernel<<<blocks, 256, 0, stream>>>(x, xq, xs, M, K);
  }
  {
    int blocks = Nb * Kb;
    quant_w_kernel<<<blocks, 256, 0, stream>>>(w, wq, wsc, N, K);
  }
  {
    dim3 grid(M / BLK, N / BLK);
    gemm_i8_kernel<<<grid, 256, 0, stream>>>(xq, xs, wq, wsc, bias, out, M, N, K);
  }
}
